// GLM4MoE_89172111000202
// MI455X (gfx1250) — compile-verified, matrix operand order checked by reading, not run on hardware
//
#include <hip/hip_runtime.h>
#include <hip/hip_bf16.h>
#include <math.h>

#define T        2048
#define HID      1024
#define INTER    512
#define NE       160
#define TOPK     8
#define RSCALE   2.5f
#define MAX_TILES 1312          // <= 16384/16 + 160 (routed) + 128 (shared)
#define MAX_ROWS  18944         // <= 16384 + 160*15, padded
#define XS_STRIDE (HID + 8)     // bf16 elems; row stride 2064 B (16B aligned)
#define H1_STRIDE (INTER + 8)   // 1040 B (16B aligned)

typedef __attribute__((ext_vector_type(16))) __bf16 v16bf;
typedef __attribute__((ext_vector_type(8)))  __bf16 v8bf;
typedef __attribute__((ext_vector_type(4)))  __bf16 v4bf;
typedef __attribute__((ext_vector_type(8)))  float  v8f;
typedef __attribute__((ext_vector_type(4)))  int    v4i;

#if defined(__gfx1250__) && __has_builtin(__builtin_amdgcn_global_load_async_to_lds_b128)
#define HAVE_ASYNC 1
typedef __attribute__((address_space(1))) v4i gv4i;   // global int4
typedef __attribute__((address_space(3))) v4i lv4i;   // LDS int4
#else
#define HAVE_ASYNC 0
#endif

__device__ __forceinline__ float fast_sigmoid(float v) {
#if defined(__gfx1250__)
  return __builtin_amdgcn_rcpf(1.f + __expf(-v));   // v_exp_f32 + v_rcp_f32
#else
  return 1.f / (1.f + __expf(-v));
#endif
}

// ---------------- Activations fp32 -> bf16 (once per launch) -------------------
__global__ __launch_bounds__(256) void cvt_x_k(const float* __restrict__ x,
                                               __bf16* __restrict__ xbf) {
  const int i = (blockIdx.x * 256 + threadIdx.x) * 4;   // T*HID/4 threads
  const float4 v = *(const float4*)(x + i);
  v4bf o;
  o[0] = (__bf16)v.x; o[1] = (__bf16)v.y; o[2] = (__bf16)v.z; o[3] = (__bf16)v.w;
  *(v4bf*)(xbf + i) = o;
}

// ---------------- Router: logits -> sigmoid -> top-8 -> weights ----------------
__global__ __launch_bounds__(256) void router_k(const float* __restrict__ x,
                                                const float* __restrict__ rw,
                                                const float* __restrict__ bias,
                                                int* __restrict__ topk_idx,
                                                float* __restrict__ topk_w,
                                                int* __restrict__ counts) {
  __shared__ float xs[HID];
  __shared__ float probs[NE];
  const int token = blockIdx.x;
  const int tid = threadIdx.x;
  for (int h = tid; h < HID; h += 256) xs[h] = x[(size_t)token * HID + h];
  __syncthreads();
  if (tid < NE) {
    const float* w = rw + (size_t)tid * HID;
    float acc = 0.f;
#pragma unroll 4
    for (int h = 0; h < HID; ++h) acc += xs[h] * w[h];
    acc += bias[tid];
    probs[tid] = fast_sigmoid(acc);
  }
  __syncthreads();
  if (tid == 0) {
    unsigned mask[5] = {0u, 0u, 0u, 0u, 0u};
    int idx[TOPK];
    float wv[TOPK];
    float sum = 0.f;
    for (int k = 0; k < TOPK; ++k) {
      float best = -1.f;
      int bi = 0;
      for (int e = 0; e < NE; ++e) {
        if ((mask[e >> 5] >> (e & 31)) & 1u) continue;
        float p = probs[e];
        if (p > best) { best = p; bi = e; }
      }
      mask[bi >> 5] |= 1u << (bi & 31);
      idx[k] = bi;
      wv[k] = best;
      sum += best;
    }
    const float inv = RSCALE / (sum + 1e-9f);
    for (int k = 0; k < TOPK; ++k) {
      topk_idx[token * TOPK + k] = idx[k];
      topk_w[token * TOPK + k] = wv[k] * inv;
      atomicAdd(&counts[idx[k]], 1);
    }
  }
}

// ------------- Scan: padded offsets + tile descriptors (+ shared tiles) --------
__global__ void scan_k(const int* __restrict__ counts, int* __restrict__ offs,
                       int* __restrict__ descE, int* __restrict__ descR,
                       int* __restrict__ nTiles) {
  if (threadIdx.x != 0 || blockIdx.x != 0) return;
  int acc = 0, nt = 0;
  for (int e = 0; e < NE; ++e) {
    offs[e] = acc;
    const int pr = (counts[e] + 15) & ~15;
    for (int t = 0; t < pr; t += 16) { descE[nt] = e; descR[nt] = acc + t; ++nt; }
    acc += pr;
  }
  offs[NE] = acc;
  for (int t = 0; t < T; t += 16) { descE[nt] = NE; descR[nt] = t; ++nt; }  // shared expert
  nTiles[0] = nt;
}

// ------------- Scatter token->expert row lists --------------------------------
__global__ __launch_bounds__(256) void scatter_k(const int* __restrict__ topk_idx,
                                                 const float* __restrict__ topk_w,
                                                 const int* __restrict__ offs,
                                                 int* __restrict__ fill,
                                                 int* __restrict__ rowTok,
                                                 float* __restrict__ rowW) {
  const int p = blockIdx.x * 256 + threadIdx.x;
  if (p >= T * TOPK) return;
  const int token = p >> 3;
  const int e = topk_idx[p];
  const float w = topk_w[p];
  const int pos = atomicAdd(&fill[e], 1);
  const int r = offs[e] + pos;
  rowTok[r] = token;
  rowW[r] = w;
}

// ------------- WMMA fragment helpers (wave32 16-bit layouts) ------------------
__device__ __forceinline__ v16bf load_a_frag(const __bf16* row, int k0, int koff) {
  const v8bf lo = *(const v8bf*)(row + k0 + koff);        // K = koff..koff+7
  const v8bf hi = *(const v8bf*)(row + k0 + 16 + koff);   // K = 16+koff..
  v16bf a;
#pragma unroll
  for (int i = 0; i < 8; ++i) { a[i] = lo[i]; a[8 + i] = hi[i]; }
  return a;
}

__device__ __forceinline__ v16bf load_b_frag(const float* col, int k0, int koff) {
  v16bf b;
#pragma unroll
  for (int i = 0; i < 8; ++i) {
    b[i]     = (__bf16)col[k0 + koff + i];
    b[8 + i] = (__bf16)col[k0 + 16 + koff + i];
  }
  return b;
}

// ------------- Grouped expert FFN: silu(X Wg^T)*w @ Wd^T ----------------------
__global__ __launch_bounds__(256) void moe_gemm_k(
    const __bf16* __restrict__ xbf, const float* __restrict__ w_gate,
    const float* __restrict__ w_down, const float* __restrict__ wg_sh,
    const float* __restrict__ wd_sh, const int* __restrict__ descE,
    const int* __restrict__ descR, const int* __restrict__ nTiles,
    const int* __restrict__ rowTok, const float* __restrict__ rowW,
    float* __restrict__ out) {
  if ((int)blockIdx.x >= nTiles[0]) return;
  const int e = descE[blockIdx.x];
  const int rowStart = descR[blockIdx.x];

  __shared__ __bf16 Xs[16 * XS_STRIDE];   // 16 x 1024 activations (bf16)
  __shared__ __bf16 H1s[16 * H1_STRIDE];  // 16 x 512 gated hidden (bf16)
  __shared__ int tok_s[16];
  __shared__ float wrow_s[16];

  const int tid = threadIdx.x;
  if (tid < 16) {
    if (e == NE) {                       // shared expert pseudo-tiles
      tok_s[tid] = rowStart + tid;
      wrow_s[tid] = 1.f;
    } else {
      const int t = rowTok[rowStart + tid];
      tok_s[tid] = t;
      wrow_s[tid] = (t >= 0) ? rowW[rowStart + tid] : 0.f;
    }
  }
  __syncthreads();

  // Stage 16 token rows (bf16) into LDS in 16-byte chunks.
  // 16 rows * (HID/8 = 128) chunks = 2048 chunks -> 8 per thread.
  for (int ch = tid; ch < 16 * (HID / 8); ch += 256) {
    const int m = ch >> 7;
    const int h8 = (ch & 127) * 8;
    __bf16* dst = &Xs[m * XS_STRIDE + h8];
    const int t = tok_s[m];
    if (t >= 0) {
#if HAVE_ASYNC
      __builtin_amdgcn_global_load_async_to_lds_b128(
          (gv4i*)(xbf + (size_t)t * HID + h8), (lv4i*)dst, 0, 0);
#else
      *(v8bf*)dst = *(const v8bf*)(xbf + (size_t)t * HID + h8);
#endif
    } else {
      v8bf z = {};
      *(v8bf*)dst = z;
    }
  }
#if HAVE_ASYNC
#if __has_builtin(__builtin_amdgcn_s_wait_asynccnt)
  __builtin_amdgcn_s_wait_asynccnt(0);
#else
  asm volatile("s_wait_asynccnt 0x0" ::: "memory");
#endif
#endif
  __syncthreads();

  const int lane = tid & 31, wave = tid >> 5;
  const int nloc = lane & 15;
  const int koff = (lane < 16) ? 0 : 8;   // K half selector (A & B frags)
  const int mbase = (lane < 16) ? 0 : 8;  // C: lanes 16-31 hold M=8..15

  // ---- GEMM1: H1 = silu(X @ Wg^T) * w_row ;  N=512 -> 4 n-tiles/wave ----
  const float* gbase = (e == NE) ? wg_sh : (w_gate + (size_t)e * INTER * HID);
#pragma unroll
  for (int nt = 0; nt < 4; ++nt) {
    const int nbase = (wave * 4 + nt) * 16;
    const float* Bp = gbase + (size_t)(nbase + nloc) * HID;
    __builtin_prefetch(Bp, 0, 0);
    v8f c = {};
    const __bf16* Arow = Xs + (lane & 15) * XS_STRIDE;
    for (int k0 = 0; k0 < HID; k0 += 32) {
      const v16bf a = load_a_frag(Arow, k0, koff);
      const v16bf b = load_b_frag(Bp, k0, koff);
      c = __builtin_amdgcn_wmma_f32_16x16x32_bf16(false, a, false, b, (short)0,
                                                  c, false, false);
    }
#pragma unroll
    for (int r = 0; r < 8; ++r) {
      const int m = mbase + r;
      const float v = c[r];
      const float s = v * fast_sigmoid(v) * wrow_s[m];
      H1s[m * H1_STRIDE + nbase + nloc] = (__bf16)s;
    }
  }
  __syncthreads();

  // ---- GEMM2: out += H1 @ Wd^T ;  N=1024 -> 8 n-tiles/wave ----
  const float* dbase = (e == NE) ? wd_sh : (w_down + (size_t)e * HID * INTER);
#pragma unroll
  for (int nt = 0; nt < 8; ++nt) {
    const int nbase = (wave * 8 + nt) * 16;
    const float* Bp = dbase + (size_t)(nbase + nloc) * INTER;
    __builtin_prefetch(Bp, 0, 0);
    v8f c = {};
    const __bf16* Arow = H1s + (lane & 15) * H1_STRIDE;
    for (int k0 = 0; k0 < INTER; k0 += 32) {
      const v16bf a = load_a_frag(Arow, k0, koff);
      const v16bf b = load_b_frag(Bp, k0, koff);
      c = __builtin_amdgcn_wmma_f32_16x16x32_bf16(false, a, false, b, (short)0,
                                                  c, false, false);
    }
#pragma unroll
    for (int r = 0; r < 8; ++r) {
      const int m = mbase + r;
      const int t = tok_s[m];
      if (t >= 0) atomicAdd(&out[(size_t)t * HID + nbase + nloc], c[r]);
    }
  }
}

// ------------------------------- launcher -------------------------------------
extern "C" void kernel_launch(void* const* d_in, const int* in_sizes, int n_in,
                              void* d_out, int out_size, void* d_ws, size_t ws_size,
                              hipStream_t stream) {
  const float* x    = (const float*)d_in[0];
  const float* rw   = (const float*)d_in[1];
  const float* bias = (const float*)d_in[2];
  const float* wg   = (const float*)d_in[3];
  const float* wd   = (const float*)d_in[4];
  const float* wgs  = (const float*)d_in[5];
  const float* wds  = (const float*)d_in[6];
  float* out = (float*)d_out;

  int*    topk_idx = (int*)d_ws;                   // T*8
  float*  topk_w   = (float*)(topk_idx + T * TOPK);// T*8
  int*    counts   = (int*)(topk_w + T * TOPK);    // NE
  int*    fill     = counts + NE;                  // NE
  int*    offs     = fill + NE;                    // NE+1
  int*    nTiles   = offs + NE + 1;                // 1
  int*    descE    = nTiles + 1;                   // MAX_TILES
  int*    descR    = descE + MAX_TILES;            // MAX_TILES
  int*    rowTok   = descR + MAX_TILES;            // MAX_ROWS
  float*  rowW     = (float*)(rowTok + MAX_ROWS);  // MAX_ROWS
  __bf16* xbf      = (__bf16*)(rowW + MAX_ROWS);   // T*HID bf16 (4 MB)

  (void)hipMemsetAsync(counts, 0, 2 * NE * sizeof(int), stream);        // counts+fill
  (void)hipMemsetAsync(rowTok, 0xFF, MAX_ROWS * sizeof(int), stream);   // rows = -1
  (void)hipMemsetAsync(d_out, 0, (size_t)out_size * sizeof(float), stream);

  cvt_x_k<<<(T * HID / 4) / 256, 256, 0, stream>>>(x, xbf);
  router_k<<<T, 256, 0, stream>>>(x, rw, bias, topk_idx, topk_w, counts);
  scan_k<<<1, 1, 0, stream>>>(counts, offs, descE, descR, nTiles);
  scatter_k<<<(T * TOPK + 255) / 256, 256, 0, stream>>>(topk_idx, topk_w, offs,
                                                        fill, rowTok, rowW);
  moe_gemm_k<<<MAX_TILES, 256, 0, stream>>>(xbf, wg, wd, wgs, wds, descE, descR,
                                            nTiles, rowTok, rowW, out);
}
